// MultipleNegativesRankingLoss_27127013442181
// MI455X (gfx1250) — matrix operand order repaired and checked by reading, not compiled, not run on hardware
//
#include <hip/hip_runtime.h>

typedef __attribute__((ext_vector_type(4)))  _Float16 v4h;
typedef __attribute__((ext_vector_type(8)))  _Float16 v8h;
typedef __attribute__((ext_vector_type(16))) _Float16 v16h;
typedef __attribute__((ext_vector_type(8)))  float    v8f;

#define B_ROWS 8192
#define DIM    1024
#define SCALE  20.0f
// SCALE * log2(e): exp(SCALE*x - SCALE) == exp2(C2*x - C2)
#define C2     28.853900817779268f
#define LBSTRIDE 40   // padded LDS row stride in halves (80 B)

// CDNA5 async global->LDS copy (ASYNCcnt), bypassing the VGPR round trip.
__device__ __forceinline__ void async_ld128(unsigned ldsoff, const _Float16* g) {
  asm volatile("global_load_async_to_lds_b128 %0, %1, off"
               :: "v"(ldsoff), "v"((unsigned long long)g)
               : "memory");
}
__device__ __forceinline__ void wait_async0() {
  asm volatile("s_wait_asynccnt 0x0" ::: "memory");
}

// ---------------------------------------------------------------------------
// Pass 1: L2-normalize rows of anchors & positives, emit fp16 to workspace.
// ---------------------------------------------------------------------------
__global__ __launch_bounds__(256) void mnrl_normalize(
    const float* __restrict__ A, const float* __restrict__ P,
    _Float16* __restrict__ ah, _Float16* __restrict__ ph) {
  const int row = blockIdx.x;
  const int tid = threadIdx.x;

  const float4 va = ((const float4*)(A + (size_t)row * DIM))[tid];
  const float4 vp = ((const float4*)(P + (size_t)row * DIM))[tid];

  float sa = va.x * va.x + va.y * va.y + va.z * va.z + va.w * va.w;
  float sp = vp.x * vp.x + vp.y * vp.y + vp.z * vp.z + vp.w * vp.w;

  #pragma unroll
  for (int m = 1; m < 32; m <<= 1) {
    sa += __shfl_xor(sa, m, 32);
    sp += __shfl_xor(sp, m, 32);
  }
  __shared__ float ra[8], rp[8];
  const int wid = tid >> 5, lane = tid & 31;
  if (lane == 0) { ra[wid] = sa; rp[wid] = sp; }
  __syncthreads();
  float ta = 0.0f, tp = 0.0f;
  #pragma unroll
  for (int i = 0; i < 8; i++) { ta += ra[i]; tp += rp[i]; }

  const float ia = 1.0f / fmaxf(sqrtf(ta), 1e-12f);
  const float ip = 1.0f / fmaxf(sqrtf(tp), 1e-12f);

  v4h oa = { (_Float16)(va.x * ia), (_Float16)(va.y * ia),
             (_Float16)(va.z * ia), (_Float16)(va.w * ia) };
  v4h op = { (_Float16)(vp.x * ip), (_Float16)(vp.y * ip),
             (_Float16)(vp.z * ip), (_Float16)(vp.w * ip) };
  *(v4h*)(ah + (size_t)row * DIM + tid * 4) = oa;
  *(v4h*)(ph + (size_t)row * DIM + tid * 4) = op;
}

// Fragment loaders ----------------------------------------------------------
__device__ __forceinline__ v16h load_afrag(const _Float16* arow, int kc, int hf) {
  // 16-bit A 16x32: half 0 -> K 0..7 & 16..23, half 1 -> K 8..15 & 24..31
  v8h lo = *(const v8h*)(arow + kc + hf * 8);
  v8h hi = *(const v8h*)(arow + kc + 16 + hf * 8);
  return __builtin_shufflevector(lo, hi, 0,1,2,3,4,5,6,7,8,9,10,11,12,13,14,15);
}
__device__ __forceinline__ v16h load_bfrag(const _Float16* buf, int t, int lm, int hf) {
  // B 32x16: lane = column, half 0 -> K 0..7 & 16..23, half 1 -> K 8..15 & 24..31
  const _Float16* bp = buf + (t * 16 + lm) * LBSTRIDE + hf * 8;
  v8h lo = *(const v8h*)(bp);
  v8h hi = *(const v8h*)(bp + 16);
  return __builtin_shufflevector(lo, hi, 0,1,2,3,4,5,6,7,8,9,10,11,12,13,14,15);
}
__device__ __forceinline__ v8f wmma(v16h a, v16h b, v8f c) {
  return __builtin_amdgcn_wmma_f32_16x16x32_f16(false, a, false, b, (short)0, c,
                                                false, false);
}

// ---------------------------------------------------------------------------
// Pass 2: fused WMMA GEMM + softmax-sum with fixed max bound.
// Logits are 20*cosine, bounded by +/-20, so exp(x - 20) can never overflow:
// no online max tracking is needed, only running sums of exp2(C2*x - C2).
// grid = (64 row-blocks of 128 rows, 8 column splits), block = 128 = 4 waves.
// Each wave owns TWO 16-row m-tiles; B chunks stream via async-to-LDS (x2 buf).
// ---------------------------------------------------------------------------
__global__ __launch_bounds__(128) void mnrl_gemm_softmax(
    const _Float16* __restrict__ ah, const _Float16* __restrict__ ph,
    float* __restrict__ pl, float* __restrict__ dg) {
  const int bm    = blockIdx.x;   // 0..63 : which 128-row block
  const int split = blockIdx.y;   // 0..7  : which group of 8 column steps
  const int tid   = threadIdx.x;
  const int wave  = tid >> 5;     // 0..3
  const int lane  = tid & 31;
  const int lm    = lane & 15;
  const int hf    = lane >> 4;
  const int m0    = bm * 128 + wave * 32;   // first row of this wave's 2 m-tiles

  __shared__ __attribute__((aligned(16))) _Float16 lb[2][128 * LBSTRIDE];

  const _Float16* arow0 = ah + (size_t)(m0 + lm) * DIM;
  const _Float16* arow1 = arow0 + (size_t)16 * DIM;

  const unsigned ldsrow0 = (unsigned)(size_t)(&lb[0][0]) + (unsigned)tid * (LBSTRIDE * 2);
  const unsigned ldsrow1 = (unsigned)(size_t)(&lb[1][0]) + (unsigned)tid * (LBSTRIDE * 2);

  float rl0[8], rl1[8];
  #pragma unroll
  for (int r = 0; r < 8; r++) { rl0[r] = 0.0f; rl1[r] = 0.0f; }

  const int dj = bm;  // column step whose 128 cols contain this block's diagonal

  for (int js = split * 8; js < split * 8 + 8; ++js) {
    const int n0 = js * 128;
    v8f acc0[8] = {};
    v8f acc1[8] = {};

    // Prologue: stream first B chunk (rows n0..n0+127, k 0..31) into buffer 0.
    {
      const _Float16* g = ph + (size_t)(n0 + tid) * DIM;
      async_ld128(ldsrow0,      g);
      async_ld128(ldsrow0 + 16, g + 8);
      async_ld128(ldsrow0 + 32, g + 16);
      async_ld128(ldsrow0 + 48, g + 24);
    }

    int phs = 0;
    for (int kc = 0; kc < DIM; kc += 32, phs ^= 1) {
      wait_async0();        // our async writes into current buffer done
      __syncthreads();      // everyone's writes done; prior reads of next buf done

      if (kc + 32 < DIM) {  // stream next chunk into the other buffer
        const _Float16* g = ph + (size_t)(n0 + tid) * DIM + (kc + 32);
        const unsigned ln = phs ? ldsrow0 : ldsrow1;
        async_ld128(ln,      g);
        async_ld128(ln + 16, g + 8);
        async_ld128(ln + 32, g + 16);
        async_ld128(ln + 48, g + 24);
      }

      const _Float16* bbuf = &lb[phs][0];
      const v16h fA0 = load_afrag(arow0, kc, hf);
      const v16h fA1 = load_afrag(arow1, kc, hf);

      // 2-stage pipeline: fetch B[t+1] while both WMMAs consume B[t].
      v16h fBc = load_bfrag(bbuf, 0, lm, hf);
      #pragma unroll
      for (int t = 0; t < 8; t++) {
        v16h fBn;
        if (t < 7) fBn = load_bfrag(bbuf, t + 1, lm, hf);
        acc0[t] = wmma(fA0, fBc, acc0[t]);
        acc1[t] = wmma(fA1, fBc, acc1[t]);
        fBc = fBn;
      }
    }

    // Accumulate sum of exp(logit - SCALE) for this 16x128 strip.
    // C layout: vgpr r, lanes 0-15 -> (M=r, N=lane); lanes 16-31 -> (M=8+r).
    auto epilogue = [&](v8f (&acc)[8], float (&rlu)[8], int u) {
      #pragma unroll
      for (int r = 0; r < 8; r++) {
        float ss = 0.0f;
        #pragma unroll
        for (int t = 0; t < 8; t++)
          ss += __builtin_amdgcn_exp2f(fmaf(C2, acc[t][r], -C2));
        ss += __shfl_xor(ss, 1, 32);
        ss += __shfl_xor(ss, 2, 32);
        ss += __shfl_xor(ss, 4, 32);
        ss += __shfl_xor(ss, 8, 32);
        rlu[r] += ss;
      }
      if (js == dj) {   // capture diagonal logits (tile t == wave*2 + u)
        #pragma unroll
        for (int t = 0; t < 8; t++) {
          if (t == wave * 2 + u) {
            const int rbase = m0 + u * 16;
            #pragma unroll
            for (int r = 0; r < 8; r++) {
              if (lane == r)      dg[rbase + r]     = SCALE * acc[t][r];
              if (lane == 24 + r) dg[rbase + 8 + r] = SCALE * acc[t][r];
            }
          }
        }
      }
    };
    epilogue(acc0, rl0, 0);
    epilogue(acc1, rl1, 1);
  }

  // One lane per half-wave writes the row partials it holds (replicated).
  if (lm == 0) {
    #pragma unroll
    for (int r = 0; r < 8; r++) {
      const int row0 = m0 + hf * 8 + r;
      const int row1 = m0 + 16 + hf * 8 + r;
      pl[row0 * 8 + split] = rl0[r];
      pl[row1 * 8 + split] = rl1[r];
    }
  }
}

// ---------------------------------------------------------------------------
// Pass 3: merge 8 sum partials per row: lse = SCALE + log(sum), subtract
// diagonal, mean -> scalar.
// ---------------------------------------------------------------------------
__global__ __launch_bounds__(256) void mnrl_finalize(
    const float* __restrict__ pl, const float* __restrict__ dg,
    float* __restrict__ out) {
  __shared__ float red[256];
  const int tid = threadIdx.x;
  float s = 0.0f;
  for (int i = tid; i < B_ROWS; i += 256) {
    float L = 0.0f;
    #pragma unroll
    for (int k = 0; k < 8; k++) L += pl[i * 8 + k];
    s += (SCALE + __logf(L)) - dg[i];
  }
  red[tid] = s;
  __syncthreads();
  for (int off = 128; off > 0; off >>= 1) {
    if (tid < off) red[tid] += red[tid + off];
    __syncthreads();
  }
  if (tid == 0) out[0] = red[0] * (1.0f / (float)B_ROWS);
}

// ---------------------------------------------------------------------------
extern "C" void kernel_launch(void* const* d_in, const int* in_sizes, int n_in,
                              void* d_out, int out_size, void* d_ws, size_t ws_size,
                              hipStream_t stream) {
  (void)in_sizes; (void)n_in; (void)out_size; (void)ws_size;

  const float* anchors   = (const float*)d_in[0];
  const float* positives = (const float*)d_in[1];
  float* out = (float*)d_out;

  char* w = (char*)d_ws;
  _Float16* ah = (_Float16*)w;                                   // 16 MB
  _Float16* ph = (_Float16*)(w + ((size_t)16 << 20));            // 16 MB
  float*    pl = (float*)(w + ((size_t)32 << 20));               // 8192*8 f32
  float*    dg = pl + (size_t)B_ROWS * 8;                        // 8192 f32

  mnrl_normalize<<<dim3(B_ROWS), dim3(256), 0, stream>>>(anchors, positives, ah, ph);

  mnrl_gemm_softmax<<<dim3(64, 8), dim3(128), 0, stream>>>(ah, ph, pl, dg);

  mnrl_finalize<<<dim3(1), dim3(256), 0, stream>>>(pl, dg, out);
}
